// SlowMultiHeadAttention_10299331576242
// MI455X (gfx1250) — matrix operand order repaired and checked by reading, not compiled, not run on hardware
//
#include <hip/hip_runtime.h>
#include <hip/hip_bf16.h>

// ---------------------------------------------------------------------------
// Multi-head attention for MI455X (gfx1250), wave32 + v_wmma_f32_16x16x32_f16
// ---------------------------------------------------------------------------

typedef _Float16 v16h __attribute__((ext_vector_type(16)));
typedef _Float16 v8h  __attribute__((ext_vector_type(8)));
typedef float    v8f  __attribute__((ext_vector_type(8)));

constexpr int B_  = 2;
constexpr int S_  = 2048;
constexpr int D_  = 1024;
constexpr int H_  = 16;
constexpr int DK_ = 64;

// ---- WMMA fragment helpers (layouts per CDNA5 ISA §7.12.2) ------

// A fragment (16x32 f16), row-major source, base points at tile origin.
// lane L: m = L%16 ; halves: k = (L/16)*8 + {0..7} and +16..+23
__device__ __forceinline__ v16h load_a_frag(const _Float16* base, int ld, int lane) {
    int m    = lane & 15;
    int koff = (lane >> 4) << 3;
    const _Float16* p = base + m * ld + koff;
    v8h lo = *(const v8h*)(p);
    v8h hi = *(const v8h*)(p + 16);
    v16h a;
#pragma unroll
    for (int i = 0; i < 8; ++i) { a[i] = lo[i]; a[i + 8] = hi[i]; }
    return a;
}

// B fragment (32x16 f16) loaded from B^T storage (N-major: [n][k], stride ldt).
// lane L: n = L%16 ; halves: k = (L/16)*8 + {0..7} and +16..+23
__device__ __forceinline__ v16h load_b_fragT(const _Float16* baseT, int ldt, int lane) {
    int n  = lane & 15;
    int kb = (lane >> 4) << 3;
    const _Float16* p = baseT + n * ldt + kb;
    v8h lo = *(const v8h*)(p);
    v8h hi = *(const v8h*)(p + 16);
    v16h b;
#pragma unroll
    for (int i = 0; i < 8; ++i) { b[i] = lo[i]; b[i + 8] = hi[i]; }
    return b;
}

__device__ __forceinline__ v8f wmma_f16(v16h a, v16h b, v8f c) {
    return __builtin_amdgcn_wmma_f32_16x16x32_f16(false, a, false, b, (short)0, c,
                                                  false, false);
}

// ---- cast kernels ----------------------------------------------------------

__global__ void cast_f32_f16_kernel(const float* __restrict__ in,
                                    _Float16* __restrict__ out, int n) {
    int i = blockIdx.x * blockDim.x + threadIdx.x;
    if (i < n) out[i] = (_Float16)in[i];
}

// in: (H, D, DK) f32  ->  out: (H, DK, D) f16
__global__ void cast_transpose_w_kernel(const float* __restrict__ in,
                                        _Float16* __restrict__ out, int n) {
    int i = blockIdx.x * blockDim.x + threadIdx.x;
    if (i >= n) return;
    int hd = i / DK_;
    int k  = i - hd * DK_;
    int h  = hd / D_;
    int d  = hd - h * D_;
    out[(h * DK_ + k) * D_ + d] = (_Float16)in[i];
}

// ---- QKV projection: Q/K (S x DK) and V stored transposed (DK x S) ---------

__global__ __launch_bounds__(32) void qkv_proj_kernel(
    const _Float16* __restrict__ xh,
    const _Float16* __restrict__ WqT,
    const _Float16* __restrict__ WkT,
    const _Float16* __restrict__ WvT,
    _Float16* __restrict__ Q,
    _Float16* __restrict__ K,
    _Float16* __restrict__ VT) {
    __shared__ __attribute__((aligned(16))) _Float16 xt[16 * D_];

    int lane = threadIdx.x;
    int st   = blockIdx.x;   // 0..B*S/16-1
    int h    = blockIdx.y;   // head
    int row0 = st * 16;      // row in flattened (B*S)

    // cooperative LDS stage of the 16 x 1024 x-tile (32 KB)
    const _Float16* src = xh + (long)row0 * D_;
#pragma unroll 4
    for (int i = 0; i < (16 * D_) / (8 * 32); ++i) {
        int f = (i * 32 + lane) * 8;
        *(v8h*)(&xt[f]) = *(const v8h*)(src + f);
    }
    asm volatile("s_wait_dscnt 0" ::: "memory");

    int  b    = row0 / S_;
    int  srow = row0 - b * S_;
    long bh   = (long)(b * H_ + h);

    const _Float16* Ws[3] = {WqT + (long)h * DK_ * D_,
                             WkT + (long)h * DK_ * D_,
                             WvT + (long)h * DK_ * D_};

    int n  = lane & 15;
    int mb = (lane >> 4) * 8;

#pragma unroll
    for (int mtx = 0; mtx < 3; ++mtx) {
        v8f acc0 = {}, acc1 = {}, acc2 = {}, acc3 = {};
        for (int ks = 0; ks < D_ / 32; ++ks) {
            v16h a = load_a_frag(&xt[ks * 32], D_, lane);
            v16h b0 = load_b_fragT(Ws[mtx] + (0 * 16) * D_ + ks * 32, D_, lane);
            v16h b1 = load_b_fragT(Ws[mtx] + (1 * 16) * D_ + ks * 32, D_, lane);
            v16h b2 = load_b_fragT(Ws[mtx] + (2 * 16) * D_ + ks * 32, D_, lane);
            v16h b3 = load_b_fragT(Ws[mtx] + (3 * 16) * D_ + ks * 32, D_, lane);
            acc0 = wmma_f16(a, b0, acc0);
            acc1 = wmma_f16(a, b1, acc1);
            acc2 = wmma_f16(a, b2, acc2);
            acc3 = wmma_f16(a, b3, acc3);
        }
        v8f acc[4] = {acc0, acc1, acc2, acc3};
        if (mtx < 2) {
            _Float16* dst = (mtx == 0 ? Q : K) + (bh * S_ + srow) * DK_;
#pragma unroll
            for (int nt = 0; nt < 4; ++nt)
#pragma unroll
                for (int r = 0; r < 8; ++r)
                    dst[(mb + r) * DK_ + nt * 16 + n] = (_Float16)acc[nt][r];
        } else {
            _Float16* dst = VT + (bh * DK_) * S_ + srow;   // [dk][s]
#pragma unroll
            for (int nt = 0; nt < 4; ++nt)
#pragma unroll
                for (int r = 0; r < 8; ++r)
                    dst[(nt * 16 + n) * (long)S_ + mb + r] = (_Float16)acc[nt][r];
        }
    }
}

// ---- flash attention per (b, h, 16-row q-tile) -----------------------------

__global__ __launch_bounds__(32) void attn_kernel(
    const _Float16* __restrict__ Q,
    const _Float16* __restrict__ K,
    const _Float16* __restrict__ VT,
    _Float16* __restrict__ concat) {
    __shared__ __attribute__((aligned(16))) _Float16 Pt[16 * 32];

    int  lane = threadIdx.x;
    int  qt   = blockIdx.x;    // 0..S/16-1
    int  h    = blockIdx.y;
    int  b    = blockIdx.z;
    long bh   = (long)(b * H_ + h);

    const _Float16* Qb = Q + (bh * S_ + qt * 16) * DK_;
    const _Float16* Kb = K + bh * S_ * DK_;
    const _Float16* Vb = VT + bh * DK_ * S_;

    // Q fragments for k-steps 0..31 and 32..63 of d_k
    v16h qa0 = load_a_frag(Qb + 0,  DK_, lane);
    v16h qa1 = load_a_frag(Qb + 32, DK_, lane);

    v8f o0 = {}, o1 = {}, o2 = {}, o3 = {};
    float rmax[8], rsum[8];
#pragma unroll
    for (int r = 0; r < 8; ++r) { rmax[r] = -1e30f; rsum[r] = 0.f; }

    int n  = lane & 15;
    int mb = (lane >> 4) * 8;
    const float scale = 0.125f;   // 1/sqrt(64)

    for (int j = 0; j < S_; j += 32) {
        // scores for key columns j..j+31 : two 16x16 C tiles
        v8f sc0 = {}, sc1 = {};
        {
            v16h kb0 = load_b_fragT(Kb + (long)(j +  0) * DK_ + 0,  DK_, lane);
            v16h kb1 = load_b_fragT(Kb + (long)(j +  0) * DK_ + 32, DK_, lane);
            v16h kb2 = load_b_fragT(Kb + (long)(j + 16) * DK_ + 0,  DK_, lane);
            v16h kb3 = load_b_fragT(Kb + (long)(j + 16) * DK_ + 32, DK_, lane);
            sc0 = wmma_f16(qa0, kb0, sc0);
            sc0 = wmma_f16(qa1, kb1, sc0);
            sc1 = wmma_f16(qa0, kb2, sc1);
            sc1 = wmma_f16(qa1, kb3, sc1);
        }

        // online softmax (rows live across 16-lane halves at fixed reg r)
        float p0[8], p1[8];
#pragma unroll
        for (int r = 0; r < 8; ++r) {
            float s0 = sc0[r] * scale;
            float s1 = sc1[r] * scale;
            float t  = fmaxf(s0, s1);
#pragma unroll
            for (int msk = 1; msk < 16; msk <<= 1)
                t = fmaxf(t, __shfl_xor(t, msk, 32));
            float newm = fmaxf(rmax[r], t);
            float corr = __expf(rmax[r] - newm);
            rmax[r] = newm;
            float e0 = __expf(s0 - newm);
            float e1 = __expf(s1 - newm);
            float rs = e0 + e1;
#pragma unroll
            for (int msk = 1; msk < 16; msk <<= 1)
                rs += __shfl_xor(rs, msk, 32);
            rsum[r] = rsum[r] * corr + rs;
            o0[r] *= corr; o1[r] *= corr; o2[r] *= corr; o3[r] *= corr;
            p0[r] = e0; p1[r] = e1;
        }

        // C-layout -> A-layout via LDS bounce (16 x 32 f16 tile)
#pragma unroll
        for (int r = 0; r < 8; ++r) {
            Pt[(mb + r) * 32 + n]      = (_Float16)p0[r];
            Pt[(mb + r) * 32 + 16 + n] = (_Float16)p1[r];
        }
        asm volatile("s_wait_dscnt 0" ::: "memory");
        v16h pa = load_a_frag(&Pt[0], 32, lane);

        // O += P * V  (B fragments contiguous thanks to V^T layout)
        {
            v16h vb0 = load_b_fragT(Vb + (0 * 16) * (long)S_ + j, S_, lane);
            v16h vb1 = load_b_fragT(Vb + (1 * 16) * (long)S_ + j, S_, lane);
            v16h vb2 = load_b_fragT(Vb + (2 * 16) * (long)S_ + j, S_, lane);
            v16h vb3 = load_b_fragT(Vb + (3 * 16) * (long)S_ + j, S_, lane);
            o0 = wmma_f16(pa, vb0, o0);
            o1 = wmma_f16(pa, vb1, o1);
            o2 = wmma_f16(pa, vb2, o2);
            o3 = wmma_f16(pa, vb3, o3);
        }
    }

    // normalize and write concat[b, s, h*DK + dk]
    v8f o[4] = {o0, o1, o2, o3};
    _Float16* dst = concat + ((long)(b * S_ + qt * 16)) * D_ + h * DK_;
#pragma unroll
    for (int nt = 0; nt < 4; ++nt)
#pragma unroll
        for (int r = 0; r < 8; ++r) {
            float v = o[nt][r] / rsum[r];
            dst[(mb + r) * (long)D_ + nt * 16 + n] = (_Float16)v;
        }
}

// ---- output projection: out = concat * Wo^T (f32 result) -------------------

__global__ __launch_bounds__(32) void out_proj_kernel(
    const _Float16* __restrict__ concat,
    const _Float16* __restrict__ WoH,   // (Dout, Din) row-major f16 == B^T
    float* __restrict__ out) {
    int lane = threadIdx.x;
    int rt   = blockIdx.x;            // row tile over B*S
    int ct   = blockIdx.y;            // 64-col tile over D
    int row0 = rt * 16;
    int c0   = ct * 64;

    const _Float16* Ab = concat + (long)row0 * D_;
    v8f acc0 = {}, acc1 = {}, acc2 = {}, acc3 = {};
    for (int ks = 0; ks < D_ / 32; ++ks) {
        v16h a  = load_a_frag(Ab + ks * 32, D_, lane);
        v16h b0 = load_b_fragT(WoH + (long)(c0 + 0 * 16) * D_ + ks * 32, D_, lane);
        v16h b1 = load_b_fragT(WoH + (long)(c0 + 1 * 16) * D_ + ks * 32, D_, lane);
        v16h b2 = load_b_fragT(WoH + (long)(c0 + 2 * 16) * D_ + ks * 32, D_, lane);
        v16h b3 = load_b_fragT(WoH + (long)(c0 + 3 * 16) * D_ + ks * 32, D_, lane);
        acc0 = wmma_f16(a, b0, acc0);
        acc1 = wmma_f16(a, b1, acc1);
        acc2 = wmma_f16(a, b2, acc2);
        acc3 = wmma_f16(a, b3, acc3);
    }
    v8f acc[4] = {acc0, acc1, acc2, acc3};
    int n  = lane & 15;
    int mb = (lane >> 4) * 8;
#pragma unroll
    for (int nt = 0; nt < 4; ++nt)
#pragma unroll
        for (int r = 0; r < 8; ++r)
            out[(long)(row0 + mb + r) * D_ + c0 + nt * 16 + n] = acc[nt][r];
}

// ---------------------------------------------------------------------------

extern "C" void kernel_launch(void* const* d_in, const int* in_sizes, int n_in,
                              void* d_out, int out_size, void* d_ws, size_t ws_size,
                              hipStream_t stream) {
    const float* x  = (const float*)d_in[0];
    const float* Wq = (const float*)d_in[1];
    const float* Wk = (const float*)d_in[2];
    const float* Wv = (const float*)d_in[3];
    const float* Wo = (const float*)d_in[4];
    float* out = (float*)d_out;

    // workspace layout (f16 elements)
    _Float16* ws = (_Float16*)d_ws;
    size_t off = 0;
    _Float16* xh  = ws + off; off += (size_t)B_ * S_ * D_;       // 4 Mi halves
    _Float16* WqT = ws + off; off += (size_t)H_ * DK_ * D_;      // 1 Mi
    _Float16* WkT = ws + off; off += (size_t)H_ * DK_ * D_;
    _Float16* WvT = ws + off; off += (size_t)H_ * DK_ * D_;
    _Float16* WoH = ws + off; off += (size_t)D_ * D_;
    _Float16* Qh  = ws + off; off += (size_t)B_ * H_ * S_ * DK_;
    _Float16* Kh  = ws + off; off += (size_t)B_ * H_ * S_ * DK_;
    _Float16* VTh = ws + off; off += (size_t)B_ * H_ * S_ * DK_; // [b][h][dk][s]
    _Float16* Ch  = ws + off; off += (size_t)B_ * S_ * D_;       // concat, f16

    // 1) casts
    int nx = B_ * S_ * D_;
    cast_f32_f16_kernel<<<nx / 256, 256, 0, stream>>>(x, xh, nx);
    int nw = H_ * D_ * DK_;
    cast_transpose_w_kernel<<<nw / 256, 256, 0, stream>>>(Wq, WqT, nw);
    cast_transpose_w_kernel<<<nw / 256, 256, 0, stream>>>(Wk, WkT, nw);
    cast_transpose_w_kernel<<<nw / 256, 256, 0, stream>>>(Wv, WvT, nw);
    int no = D_ * D_;
    cast_f32_f16_kernel<<<no / 256, 256, 0, stream>>>(Wo, WoH, no);

    // 2) QKV projections (WMMA)
    dim3 gp(B_ * S_ / 16, H_);
    qkv_proj_kernel<<<gp, 32, 0, stream>>>(xh, WqT, WkT, WvT, Qh, Kh, VTh);

    // 3) flash attention (WMMA + online softmax)
    dim3 ga(S_ / 16, H_, B_);
    attn_kernel<<<ga, 32, 0, stream>>>(Qh, Kh, VTh, Ch);

    // 4) output projection (WMMA, f32 out)
    dim3 go(B_ * S_ / 16, D_ / 64);
    out_proj_kernel<<<go, 32, 0, stream>>>(Ch, WoH, out);
}
